// NeuralSpline_75290776698994
// MI455X (gfx1250) — matrix operand order repaired and checked by reading, not compiled, not run on hardware
//
#include <hip/hip_runtime.h>
#include <hip/hip_bf16.h>

typedef __attribute__((ext_vector_type(16))) _Float16 v16h;
typedef __attribute__((ext_vector_type(8)))  float    v8f;
typedef __attribute__((ext_vector_type(4)))  float    v4f;

#define N_KNOTS 10
#define STEP_INV 9.0f
#define STEP_F (1.0f / 9.0f)
#define BN_EPS 1e-5f

// ---------------------------------------------------------------------------
// 1) Bilinear-ish 4x downsample: (8,3,1024,1024) -> (8,3,256,256)
// ---------------------------------------------------------------------------
__global__ void ds4_kernel(const float* __restrict__ x, float* __restrict__ y) {
    int idx = blockIdx.x * blockDim.x + threadIdx.x;          // over 8*3*256*256
    if (idx >= 8 * 3 * 256 * 256) return;
    int ox = idx & 255;
    int oy = (idx >> 8) & 255;
    int pc = idx >> 16;                                        // plane = b*3+c
    size_t base = ((size_t)pc * 1024 + oy * 4) * 1024 + ox * 4;
    float s = 0.f;
#pragma unroll
    for (int r = 0; r < 4; ++r) {
        v4f v = *(const v4f*)(x + base + (size_t)r * 1024);
        s += v.x + v.y + v.z + v.w;
    }
    y[idx] = s * (1.0f / 16.0f);
}

// ---------------------------------------------------------------------------
// 2) conv1 direct: 3->8 ch, 3x3 stride2, 256 -> 127, + bias + relu
// ---------------------------------------------------------------------------
__global__ void conv1_kernel(const float* __restrict__ in, const float* __restrict__ w,
                             const float* __restrict__ bias, float* __restrict__ out) {
    const int OH = 127, OW = 127, IH = 256, IW = 256;
    int idx = blockIdx.x * blockDim.x + threadIdx.x;          // 8*8*127*127
    if (idx >= 8 * 8 * OH * OW) return;
    int m  = idx % (OH * OW);
    int co = (idx / (OH * OW)) & 7;
    int b  = idx / (8 * OH * OW);
    int oy = m / OW, ox = m % OW;
    const float* ib = in + (size_t)b * 3 * IH * IW;
    float acc = bias[co];
#pragma unroll
    for (int ci = 0; ci < 3; ++ci)
#pragma unroll
        for (int ky = 0; ky < 3; ++ky)
#pragma unroll
            for (int kx = 0; kx < 3; ++kx)
                acc += ib[(ci * IH + oy * 2 + ky) * IW + ox * 2 + kx] *
                       w[((co * 3 + ci) * 3 + ky) * 3 + kx];
    out[idx] = acc > 0.f ? acc : 0.f;
}

// ---------------------------------------------------------------------------
// 3) Implicit-GEMM conv via V_WMMA_F32_16X16X32_F16, fully compile-time dims.
//    D[M=OH*OW, N=Cout] = im2col(in)[M, K=Cin*9] x W^T[K, N], fp32 accum.
//    One wave per 16x16 output tile; K-loop fully unrolled -> back-to-back
//    v_wmma with constant-offset global gathers.  A/B per CDNA5 ISA 7.12.2.
// ---------------------------------------------------------------------------
template <int Cin, int IH, int IW, int Cout, int OH, int OW>
__global__ __launch_bounds__(32) void conv_wmma_kernel(
        const float* __restrict__ in, const float* __restrict__ w,
        const float* __restrict__ bias, float* __restrict__ out) {
    constexpr int K = Cin * 9;
    constexpr int M = OH * OW;
    constexpr int tilesM = (M + 15) >> 4;
    constexpr int tilesN = Cout >> 4;
    int wid = blockIdx.x;
    int tn  = wid % tilesN; wid /= tilesN;
    int tm  = wid % tilesM;
    int b   = wid / tilesM;

    int lane = threadIdx.x;
    int kh   = lane >> 4;          // lane-half: selects K sub-block per ISA layout
    int l15  = lane & 15;
    int mrow = tm * 16 + l15;      // A row owned by this lane
    int ncol = tn * 16 + l15;      // B column owned by this lane
    bool mvalid = (mrow < M);
    int oy = mvalid ? (mrow / OW) : 0;
    int ox = mvalid ? (mrow % OW) : 0;
    // per-lane im2col base: all K offsets are compile-time constants on top
    const float* ib = in + (size_t)b * Cin * IH * IW + (oy * 2) * IW + (ox * 2);
    const float* wp = w + (size_t)ncol * K;

    v8f acc = {};
#pragma unroll
    for (int kb = 0; kb < K; kb += 32) {
        v16h av, bv;
#pragma unroll
        for (int e = 0; e < 16; ++e) {
            // A (16x32 f16): lanes0-15 row M hold K={0..7,16..23}; lanes16-31 {8..15,24..31}
            int ka = kb + ((e >> 3) << 4) + (kh << 3) + (e & 7);
            float a = 0.f;
            if (ka < K && mvalid) {
                int ci = ka / 9, r = ka - ci * 9;
                int ky = r / 3, kx = r - ky * 3;
                a = ib[(ci * IH + ky) * IW + kx];     // constant offset per e
            }
            av[e] = (_Float16)a;
            // B (32x16 f16): lanes0-15 col N hold K={0..7,16..23}; lanes16-31 {8..15,24..31}
            int kw = kb + ((e >> 3) << 4) + (kh << 3) + (e & 7);
            float bw = (kw < K) ? wp[kw] : 0.f;       // constant offset per e
            bv[e] = (_Float16)bw;
        }
        acc = __builtin_amdgcn_wmma_f32_16x16x32_f16(
            /*neg_a=*/false, av, /*neg_b=*/false, bv,
            /*c_mod=*/(short)0, acc, /*reuse_a=*/false, /*reuse_b=*/false);
    }
    // D layout: lane L: N = L&15, VGPR r -> M = r + 8*(L>=16)
    float bsc = bias[ncol];
#pragma unroll
    for (int r = 0; r < 8; ++r) {
        int m = tm * 16 + r + (kh << 3);
        if (m < M) {
            float v = acc[r] + bsc;
            out[((size_t)b * Cout + ncol) * M + m] = v > 0.f ? v : 0.f;
        }
    }
}

// ---------------------------------------------------------------------------
// 4) BatchNorm: per-channel mean / rsqrt(var+eps) over (B, H*W), then apply.
// ---------------------------------------------------------------------------
__global__ void bn_stats_kernel(const float* __restrict__ x, float* __restrict__ mr,
                                int B, int C, int M) {
    int c = blockIdx.x;
    float s = 0.f, s2 = 0.f;
    int n = B * M;
    for (int i = threadIdx.x; i < n; i += blockDim.x) {
        int b = i / M, m = i - b * M;
        float v = x[((size_t)b * C + c) * M + m];
        s += v; s2 += v * v;
    }
    __shared__ float ls[256], lq[256];
    ls[threadIdx.x] = s; lq[threadIdx.x] = s2;
    __syncthreads();
    for (int st = 128; st > 0; st >>= 1) {
        if ((int)threadIdx.x < st) {
            ls[threadIdx.x] += ls[threadIdx.x + st];
            lq[threadIdx.x] += lq[threadIdx.x + st];
        }
        __syncthreads();
    }
    if (threadIdx.x == 0) {
        float inv = 1.0f / (float)n;
        float mu = ls[0] * inv;
        float var = lq[0] * inv - mu * mu;
        mr[c]     = mu;
        mr[C + c] = rsqrtf(var + BN_EPS);
    }
}

__global__ void bn_apply_kernel(float* __restrict__ x, const float* __restrict__ mr,
                                const float* __restrict__ g, const float* __restrict__ be,
                                int B, int C, int M) {
    int idx = blockIdx.x * blockDim.x + threadIdx.x;
    int tot = B * C * M;
    if (idx >= tot) return;
    int c = (idx / M) % C;
    x[idx] = g[c] * (x[idx] - mr[c]) * mr[C + c] + be[c];
}

// ---------------------------------------------------------------------------
// 5) Global average pool (8,128,7,7) -> (8,128)
// ---------------------------------------------------------------------------
__global__ void avgpool_kernel(const float* __restrict__ x, float* __restrict__ y) {
    int t = blockIdx.x * blockDim.x + threadIdx.x;
    if (t >= 8 * 128) return;
    float s = 0.f;
    const float* p = x + (size_t)t * 49;
#pragma unroll
    for (int k = 0; k < 49; ++k) s += p[k];
    y[t] = s * (1.0f / 49.0f);
}

// ---------------------------------------------------------------------------
// 6) FC layers (tiny: 8x128 @ 128x128^T, 8x128 @ 30x128^T)
// ---------------------------------------------------------------------------
__global__ void fc1_kernel(const float* __restrict__ h, const float* __restrict__ w,
                           const float* __restrict__ bias, float* __restrict__ y) {
    int t = blockIdx.x * blockDim.x + threadIdx.x;
    if (t >= 8 * 128) return;
    int b = t >> 7, j = t & 127;
    float s = bias[j];
    for (int k = 0; k < 128; ++k) s += h[b * 128 + k] * w[j * 128 + k];
    y[t] = s > 0.f ? s : 0.f;
}

__global__ void fc2_kernel(const float* __restrict__ h, const float* __restrict__ w,
                           const float* __restrict__ bias, float* __restrict__ y) {
    int t = blockIdx.x * blockDim.x + threadIdx.x;
    if (t >= 8 * 30) return;
    int b = t / 30, j = t - b * 30;
    float s = bias[j];
    for (int k = 0; k < 128; ++k) s += h[b * 128 + k] * w[j * 128 + k];
    y[t] = s;
}

// ---------------------------------------------------------------------------
// 7) Natural cubic spline coefficients (Thomas solve == mat @ ys of reference)
//    One thread per (b, channel).  coef[t*36]: a[0..8] b[9..17] c[18..26] d[27..35]
// ---------------------------------------------------------------------------
__global__ void spline_coef_kernel(const float* __restrict__ ys, float* __restrict__ coef) {
    int t = blockIdx.x * blockDim.x + threadIdx.x;
    if (t >= 24) return;
    int b = t / 3, c = t - b * 3;
    float y[N_KNOTS];
#pragma unroll
    for (int i = 0; i < N_KNOTS; ++i)
        y[i] = ys[b * 30 + c * 10 + i] + (float)i * STEP_F;
    // interior second derivatives: M[i-1] + 4 M[i] + M[i+1] = 6*d2y/h^2, M[0]=M[9]=0
    float rhs[9], cp[9], dp[9], Mv[N_KNOTS];
#pragma unroll
    for (int i = 1; i <= 8; ++i)
        rhs[i - 1] = 6.0f * (y[i - 1] - 2.0f * y[i] + y[i + 1]) * (STEP_INV * STEP_INV);
    cp[0] = 0.25f; dp[0] = rhs[0] * 0.25f;
#pragma unroll
    for (int i = 1; i < 8; ++i) {
        float mlt = 4.0f - cp[i - 1];
        cp[i] = 1.0f / mlt;
        dp[i] = (rhs[i] - dp[i - 1]) / mlt;
    }
    Mv[0] = 0.f; Mv[9] = 0.f;
    Mv[8] = dp[7];
#pragma unroll
    for (int i = 6; i >= 0; --i) Mv[i + 1] = dp[i] - cp[i] * Mv[i + 2];
    float* o = coef + t * 36;
#pragma unroll
    for (int i = 0; i < 9; ++i) {
        o[i]      = (Mv[i + 1] - Mv[i]) * (STEP_INV / 6.0f);            // a
        o[9 + i]  = Mv[i] * 0.5f;                                       // b
        o[18 + i] = (y[i + 1] - y[i]) * STEP_INV
                    - (Mv[i + 1] + 2.0f * Mv[i]) * (STEP_F / 6.0f);     // c
        o[27 + i] = y[i];                                               // d
    }
}

__device__ __forceinline__ float spline_eval(const float* lc, float v) {
    float xi = floorf(v * STEP_INV);
    xi = fminf(fmaxf(xi, 0.f), 8.f);
    float xf = v - xi * STEP_F;
    int k = (int)xi;
    return ((lc[k] * xf + lc[9 + k]) * xf + lc[18 + k]) * xf + lc[27 + k];
}

// ---------------------------------------------------------------------------
// 8) HEAVY KERNEL: apply spline to every pixel.  100 MB in + 100 MB out,
//    bandwidth-bound -> float4 nontemporal streaming; coeffs via LDS gather.
// ---------------------------------------------------------------------------
__global__ __launch_bounds__(256) void spline_apply_kernel(
        const float* __restrict__ x, const float* __restrict__ coef,
        float* __restrict__ out) {
    const int HW4 = (1024 * 1024) / 4;
    int plane = blockIdx.y;                         // 0..23 == b*3+c
    __shared__ float lc[36];
    if (threadIdx.x < 36) lc[threadIdx.x] = coef[plane * 36 + threadIdx.x];
    __syncthreads();
    const v4f* xin = (const v4f*)x + (size_t)plane * HW4;
    v4f*       o   = (v4f*)out     + (size_t)plane * HW4;
    for (int i = blockIdx.x * blockDim.x + threadIdx.x; i < HW4;
         i += gridDim.x * blockDim.x) {
        v4f v = __builtin_nontemporal_load(xin + i);
        v4f r;
        r.x = spline_eval(lc, v.x);
        r.y = spline_eval(lc, v.y);
        r.z = spline_eval(lc, v.z);
        r.w = spline_eval(lc, v.w);
        __builtin_nontemporal_store(r, o + i);
    }
}

// 9) Curve samples: splines output (1,8,3,255) appended after `out`.
__global__ void spline_curves_kernel(const float* __restrict__ coef, float* __restrict__ out) {
    int plane = blockIdx.x;                         // 24 planes
    int i = threadIdx.x;
    if (i >= 255) return;
    const float* lc = coef + plane * 36;
    out[plane * 255 + i] = spline_eval(lc, (float)i * (1.0f / 255.0f));
}

// ---------------------------------------------------------------------------
extern "C" void kernel_launch(void* const* d_in, const int* in_sizes, int n_in,
                              void* d_out, int out_size, void* d_ws, size_t ws_size,
                              hipStream_t stream) {
    const float* x     = (const float*)d_in[0];
    const float* w1    = (const float*)d_in[1];
    const float* b1    = (const float*)d_in[2];
    const float* w2    = (const float*)d_in[3];
    const float* b2    = (const float*)d_in[4];
    const float* g2    = (const float*)d_in[5];
    const float* be2   = (const float*)d_in[6];
    const float* w3    = (const float*)d_in[7];
    const float* b3    = (const float*)d_in[8];
    const float* g3    = (const float*)d_in[9];
    const float* be3   = (const float*)d_in[10];
    const float* w4    = (const float*)d_in[11];
    const float* b4    = (const float*)d_in[12];
    const float* g4    = (const float*)d_in[13];
    const float* be4   = (const float*)d_in[14];
    const float* w5    = (const float*)d_in[15];
    const float* b5    = (const float*)d_in[16];
    const float* g5    = (const float*)d_in[17];
    const float* be5   = (const float*)d_in[18];
    const float* fc1w  = (const float*)d_in[19];
    const float* fc1b  = (const float*)d_in[20];
    const float* fc2w  = (const float*)d_in[21];
    const float* fc2b  = (const float*)d_in[22];
    float* out = (float*)d_out;

    // workspace layout (floats)
    float* ws     = (float*)d_ws;
    float* smallb = ws;                       // 8*3*256*256   = 1572864
    float* c1     = smallb + 1572864;         // 8*8*127*127   = 1032256
    float* c2     = c1 + 1032256;             // 8*16*63*63    =  508032
    float* c3     = c2 + 508032;              // 8*32*31*31    =  246016
    float* c4     = c3 + 246016;              // 8*64*15*15    =  115200
    float* c5     = c4 + 115200;              // 8*128*7*7     =   50176
    float* stats  = c5 + 50176;               // 2*128 (reused per layer)
    float* pooled = stats + 256;              // 8*128
    float* h1     = pooled + 1024;            // 8*128
    float* ysb    = h1 + 1024;                // 8*30
    float* coef   = ysb + 240;                // 24*36

    const int B = 8;

    // 1) downsample
    ds4_kernel<<<(8 * 3 * 256 * 256 + 255) / 256, 256, 0, stream>>>(x, smallb);

    // 2) conv1 (direct)
    conv1_kernel<<<(8 * 8 * 127 * 127 + 255) / 256, 256, 0, stream>>>(smallb, w1, b1, c1);

    // 3) conv2 (WMMA) + BN : 8ch 127x127 -> 16ch 63x63
    {
        constexpr int M = 63 * 63, C = 16;
        conv_wmma_kernel<8, 127, 127, 16, 63, 63>
            <<<B * ((M + 15) / 16) * (C / 16), 32, 0, stream>>>(c1, w2, b2, c2);
        bn_stats_kernel<<<C, 256, 0, stream>>>(c2, stats, B, C, M);
        bn_apply_kernel<<<(B * C * M + 255) / 256, 256, 0, stream>>>(c2, stats, g2, be2, B, C, M);
    }
    // 4) conv3 (WMMA) + BN : 16ch 63x63 -> 32ch 31x31
    {
        constexpr int M = 31 * 31, C = 32;
        conv_wmma_kernel<16, 63, 63, 32, 31, 31>
            <<<B * ((M + 15) / 16) * (C / 16), 32, 0, stream>>>(c2, w3, b3, c3);
        bn_stats_kernel<<<C, 256, 0, stream>>>(c3, stats, B, C, M);
        bn_apply_kernel<<<(B * C * M + 255) / 256, 256, 0, stream>>>(c3, stats, g3, be3, B, C, M);
    }
    // 5) conv4 (WMMA) + BN : 32ch 31x31 -> 64ch 15x15
    {
        constexpr int M = 15 * 15, C = 64;
        conv_wmma_kernel<32, 31, 31, 64, 15, 15>
            <<<B * ((M + 15) / 16) * (C / 16), 32, 0, stream>>>(c3, w4, b4, c4);
        bn_stats_kernel<<<C, 256, 0, stream>>>(c4, stats, B, C, M);
        bn_apply_kernel<<<(B * C * M + 255) / 256, 256, 0, stream>>>(c4, stats, g4, be4, B, C, M);
    }
    // 6) conv5 (WMMA) + BN : 64ch 15x15 -> 128ch 7x7
    {
        constexpr int M = 7 * 7, C = 128;
        conv_wmma_kernel<64, 15, 15, 128, 7, 7>
            <<<B * ((M + 15) / 16) * (C / 16), 32, 0, stream>>>(c4, w5, b5, c5);
        bn_stats_kernel<<<C, 256, 0, stream>>>(c5, stats, B, C, M);
        bn_apply_kernel<<<(B * C * M + 255) / 256, 256, 0, stream>>>(c5, stats, g5, be5, B, C, M);
    }

    // 7) pool + FCs + knots
    avgpool_kernel<<<(8 * 128 + 255) / 256, 256, 0, stream>>>(c5, pooled);
    fc1_kernel<<<4, 256, 0, stream>>>(pooled, fc1w, fc1b, h1);
    fc2_kernel<<<1, 256, 0, stream>>>(h1, fc2w, fc2b, ysb);
    spline_coef_kernel<<<1, 32, 0, stream>>>(ysb, coef);

    // 8) heavy streaming apply: (1,8,3,1024,1024) flat == (b,c,pix)
    spline_apply_kernel<<<dim3(1024, 24), 256, 0, stream>>>(x, coef, out);

    // 9) curve samples appended at out + 8*3*1024*1024
    spline_curves_kernel<<<24, 256, 0, stream>>>(coef, out + (size_t)8 * 3 * 1024 * 1024);
}
